// Meta3D_53042846106124
// MI455X (gfx1250) — matrix operand order repaired and checked by reading, not compiled, not run on hardware
//
#include <hip/hip_runtime.h>
#include <hip/hip_bf16.h>

typedef __attribute__((ext_vector_type(16))) __bf16 v16bf;
typedef __attribute__((ext_vector_type(8)))  float  v8f;

#define DEVINL __device__ __forceinline__

constexpr int Bn = 1024, Nt = 49, C = 512, NH = 8, KD = 32, DV = 128;
constexpr int DH = 1024, HQKV = 1536, MH = 2048;
constexpr int Mrows = Bn * Nt;          // 50176 (divisible by 128)
constexpr int NPAD = 64;                // 49 tokens padded to 64 for WMMA
constexpr int NOFF = 49;                // bias table offsets (7x7)
constexpr float SCALE = 0.17677669529663687f; // 32^-0.5

// native f32 -> bf16 (RNE) conversion; single v_cvt on gfx1250
DEVINL __bf16 f2bf(float f) { return (__bf16)f; }

union Frag { v16bf v; uint4 u[2]; };

DEVINL v8f wmma_bf16(v16bf a, v16bf b, v8f c) {
  return __builtin_amdgcn_wmma_f32_16x16x32_bf16(false, a, false, b,
                                                 (short)0, c, false, false);
}

// 16x32 bf16 A fragment, row-major A. Lanes 0-15: M=lane, K={0..7, 16..23};
// lanes 16-31: M=lane-16, K={8..15, 24..31} (ISA 7.12.2 16-bit A layout).
DEVINL v16bf load_a(const __bf16* A, int lda, int m0, int kb, int lane) {
  const __bf16* base = A + (size_t)(m0 + (lane & 15)) * lda + kb;
  int h = (lane >> 4) * 8;
  Frag f;
  f.u[0] = *(const uint4*)(base + h);
  f.u[1] = *(const uint4*)(base + 16 + h);
  return f.v;
}

// 32x16 bf16 B fragment from K-major storage BT[n][k] (ldb = K).
// Lanes 0-15: N=lane, K={0..7, 16..23}; lanes 16-31: N=lane-16, K={8..15, 24..31}.
DEVINL v16bf load_bT(const __bf16* BT, int ldb, int kb, int n0, int lane) {
  const __bf16* base = BT + (size_t)(n0 + (lane & 15)) * ldb + kb + (lane >> 4) * 8;
  Frag f;
  f.u[0] = *(const uint4*)(base);
  f.u[1] = *(const uint4*)(base + 16);
  return f.v;
}

// ---------------- generic WMMA GEMM: 128x64 tile per WG (8 waves in M) -----
template <int K, class Epi>
__global__ __launch_bounds__(256) void wmma_gemm(const __bf16* __restrict__ A,
                                                 const __bf16* __restrict__ BT,
                                                 Epi epi) {
  int lane = threadIdx.x & 31;
  int wave = threadIdx.x >> 5;
  int m0 = blockIdx.x * 128 + wave * 16;
  int n0 = blockIdx.y * 64;
  v8f acc[4] = {};
  for (int kb = 0; kb < K; kb += 32) {
    v16bf a = load_a(A, K, m0, kb, lane);
#pragma unroll
    for (int t = 0; t < 4; ++t) {
      v16bf b = load_bT(BT, K, kb, n0 + 16 * t, lane);
      acc[t] = wmma_bf16(a, b, acc[t]);
    }
  }
  // C/D layout: VGPR i -> row i (lanes 0-15) / row i+8 (lanes 16-31), N=lane&15
  int rbase = m0 + ((lane >> 4) << 3);
  int cbase = n0 + (lane & 15);
#pragma unroll
  for (int t = 0; t < 4; ++t)
#pragma unroll
    for (int i = 0; i < 8; ++i)
      epi(rbase + i, cbase + 16 * t, acc[t][i]);
}

// ---------------- epilogues ------------------------------------------------
struct EpiQKV {  // +bias, split per head, q pre-scaled, v transposed
  const float* bias; __bf16* q; __bf16* k; __bf16* vT;
  DEVINL void operator()(int m, int c, float v) const {
    float val = v + bias[c];
    int b = m / Nt, n = m - b * Nt;
    int h = c / 192, w = c - h * 192;
    int bh = b * NH + h;
    if (w < KD)        q[((size_t)bh * NPAD + n) * KD + w] = f2bf(val * SCALE);
    else if (w < 2*KD) k[((size_t)bh * NPAD + n) * KD + (w - KD)] = f2bf(val);
    else               vT[((size_t)bh * DV + (w - 2 * KD)) * NPAD + n] = f2bf(val);
  }
};
struct EpiProj {  // x1 = x + (o@Wp + b)*ls1
  const float* bias; const float* ls; const float* xres; float* x1;
  DEVINL void operator()(int m, int c, float v) const {
    size_t idx = (size_t)m * C + c;
    x1[idx] = xres[idx] + (v + bias[c]) * ls[c];
  }
};
struct EpiGelu {  // exact gelu -> bf16 hidden
  const float* bias; __bf16* out;
  DEVINL void operator()(int m, int c, float v) const {
    float t = v + bias[c];
    float g = 0.5f * t * (1.0f + erff(t * 0.70710678118654752f));
    out[(size_t)m * MH + c] = f2bf(g);
  }
};
struct EpiOut {  // out = x1 + (hid@W2 + b)*ls2
  const float* bias; const float* ls; const float* x1; float* out;
  DEVINL void operator()(int m, int c, float v) const {
    size_t idx = (size_t)m * C + c;
    out[idx] = x1[idx] + (v + bias[c]) * ls[c];
  }
};

// ---------------- LayerNorm + cast to bf16 (one wave32 per 512-wide row) ---
__global__ __launch_bounds__(256) void ln_cast_kernel(const float* __restrict__ x,
                                                      const float* __restrict__ g,
                                                      const float* __restrict__ b,
                                                      __bf16* __restrict__ out) {
  int lane = threadIdx.x & 31;
  int wave = threadIdx.x >> 5;
  int row = blockIdx.x * 8 + wave;
  const float* xr = x + (size_t)row * C;
  float v[16];
  float s = 0.f;
#pragma unroll
  for (int j = 0; j < 16; ++j) { v[j] = xr[lane + 32 * j]; s += v[j]; }
#pragma unroll
  for (int mk = 16; mk >= 1; mk >>= 1) s += __shfl_xor(s, mk, 32);
  float mu = s * (1.0f / C);
  float q = 0.f;
#pragma unroll
  for (int j = 0; j < 16; ++j) { float d = v[j] - mu; q += d * d; }
#pragma unroll
  for (int mk = 16; mk >= 1; mk >>= 1) q += __shfl_xor(q, mk, 32);
  float r = rsqrtf(q * (1.0f / C) + 1e-5f);
  __bf16* orow = out + (size_t)row * C;
#pragma unroll
  for (int j = 0; j < 16; ++j) {
    int cc = lane + 32 * j;
    orow[cc] = f2bf((v[j] - mu) * r * g[cc] + b[cc]);
  }
}

// ---------------- weight cast + transpose to K-major -----------------------
__global__ __launch_bounds__(256) void castT_kernel(const float* __restrict__ W,
                                                    __bf16* __restrict__ WT,
                                                    int K, int Nc) {
  int idx = blockIdx.x * 256 + threadIdx.x;
  if (idx >= K * Nc) return;
  int k = idx / Nc, n = idx - k * Nc;
  WT[(size_t)n * K + k] = f2bf(W[idx]);
}

__global__ __launch_bounds__(256) void zero_kernel(uint4* p, int n) {
  int i = blockIdx.x * 256 + threadIdx.x;
  if (i < n) p[i] = make_uint4(0u, 0u, 0u, 0u);
}

// Expand (8,49) bias table through (49,49) index map into a padded, masked
// table stored directly in WMMA C-fragment order:
//   bias_full[((h*4 + wave)*32 + lane)*32 + t*8 + i]
// where r = wave*16 + (lane>=16)*8 + i, c = t*16 + (lane&15).
// Column pad = -1e30 (softmax mask), row pad = 0. Each attention lane then
// reads its 32 bias values as 8 contiguous b128 loads.
__global__ __launch_bounds__(256) void bias_expand_kernel(
    const float* __restrict__ bias_tab, const int* __restrict__ idxs,
    float* __restrict__ bias_full) {
  int t = blockIdx.x * 256 + threadIdx.x;
  if (t >= NH * NPAD * NPAD) return;
  int h = t >> 12;
  int rc = t & 4095;
  int r = rc >> 6, c = rc & 63;
  float v;
  if (c >= Nt)      v = -1e30f;
  else if (r >= Nt) v = 0.f;
  else              v = bias_tab[h * NOFF + idxs[r * Nt + c]];
  int wave = r >> 4, rr = r & 15;
  int half = rr >> 3, i = rr & 7;
  int tt = c >> 4, cl = c & 15;
  int lane = half * 16 + cl;
  bias_full[(((h * 4 + wave) * 32 + lane) << 5) + tt * 8 + i] = v;
}

// ---------------- fused attention: one WG (4 waves) per (b,h) --------------
__global__ __launch_bounds__(128) void attn_kernel(const __bf16* __restrict__ q,
                                                   const __bf16* __restrict__ k,
                                                   const __bf16* __restrict__ vT,
                                                   const float* __restrict__ bias_full,
                                                   __bf16* __restrict__ o) {
  __shared__ __bf16 P[NPAD][NPAD];  // 8 KB softmax probs, bf16
  int lane = threadIdx.x & 31;
  int wave = threadIdx.x >> 5;
  int bh = blockIdx.x;
  int b = bh >> 3, h = bh & 7;
  const __bf16* qb = q + (size_t)bh * NPAD * KD;
  const __bf16* kp = k + (size_t)bh * NPAD * KD;
  const __bf16* vb = vT + (size_t)bh * DV * NPAD;
  int m0 = wave * 16;
  int half8 = (lane >> 4) << 3;
  int cl = lane & 15;

  // S = (q*SCALE) k^T : K=32, single WMMA step per 16x16 tile
  v16bf af = load_a(qb, KD, m0, 0, lane);
  v8f s[4];
#pragma unroll
  for (int t = 0; t < 4; ++t) {
    v16bf bf_ = load_bT(kp, KD, 0, 16 * t, lane);
    v8f z = {};
    s[t] = wmma_bf16(af, bf_, z);
  }
  // + pre-expanded, pre-masked bias, already in fragment order: 8x b128 loads
  const float* bfrag = bias_full + ((((size_t)h * 4 + wave) * 32 + lane) << 5);
  float bb[32];
#pragma unroll
  for (int j = 0; j < 8; ++j)
    *(float4*)&bb[4 * j] = ((const float4*)bfrag)[j];
#pragma unroll
  for (int t = 0; t < 4; ++t)
#pragma unroll
    for (int i = 0; i < 8; ++i)
      s[t][i] = s[t][i] + bb[t * 8 + i];
  // softmax: each row lives in a 16-lane group, 4 values per lane
#pragma unroll
  for (int i = 0; i < 8; ++i) {
    float mx = fmaxf(fmaxf(s[0][i], s[1][i]), fmaxf(s[2][i], s[3][i]));
#pragma unroll
    for (int mk = 8; mk >= 1; mk >>= 1) mx = fmaxf(mx, __shfl_xor(mx, mk, 32));
    float e0 = __expf(s[0][i] - mx), e1 = __expf(s[1][i] - mx);
    float e2 = __expf(s[2][i] - mx), e3 = __expf(s[3][i] - mx);
    float sum = e0 + e1 + e2 + e3;
#pragma unroll
    for (int mk = 8; mk >= 1; mk >>= 1) sum += __shfl_xor(sum, mk, 32);
    float inv = 1.0f / sum;
    int r = m0 + half8 + i;
    P[r][cl]      = f2bf(e0 * inv);
    P[r][16 + cl] = f2bf(e1 * inv);
    P[r][32 + cl] = f2bf(e2 * inv);
    P[r][48 + cl] = f2bf(e3 * inv);
  }
  __syncthreads();

  // O = P(16x64 slab) x V(64x128): 2 K-steps x 8 d-tiles per wave
  int rb = m0 + half8;
#pragma unroll
  for (int t2 = 0; t2 < 8; ++t2) {
    v8f oacc = {};
#pragma unroll
    for (int ks = 0; ks < 2; ++ks) {
      const __bf16* abase = &P[m0 + cl][ks * 32];
      Frag fa;
      fa.u[0] = *(const uint4*)(abase + half8);
      fa.u[1] = *(const uint4*)(abase + 16 + half8);
      v16bf bfv = load_bT(vb, NPAD, ks * 32, 16 * t2, lane);
      oacc = wmma_bf16(fa.v, bfv, oacc);
    }
#pragma unroll
    for (int i = 0; i < 8; ++i) {
      int r = rb + i;
      if (r < Nt)
        o[(size_t)(b * Nt + r) * DH + h * DV + 16 * t2 + cl] = f2bf(oacc[i]);
    }
  }
}

// ---------------- launcher -------------------------------------------------
extern "C" void kernel_launch(void* const* d_in, const int* in_sizes, int n_in,
                              void* d_out, int out_size, void* d_ws, size_t ws_size,
                              hipStream_t stream) {
  const float* x        = (const float*)d_in[0];
  const float* ln1_g    = (const float*)d_in[1];
  const float* ln1_b    = (const float*)d_in[2];
  const float* W_qkv    = (const float*)d_in[3];
  const float* b_qkv    = (const float*)d_in[4];
  const float* bias_tab = (const float*)d_in[5];
  const float* W_proj   = (const float*)d_in[6];
  const float* b_proj   = (const float*)d_in[7];
  const float* ls1      = (const float*)d_in[8];
  const float* ln2_g    = (const float*)d_in[9];
  const float* ln2_b    = (const float*)d_in[10];
  const float* W1       = (const float*)d_in[11];
  const float* b1       = (const float*)d_in[12];
  const float* W2       = (const float*)d_in[13];
  const float* b2       = (const float*)d_in[14];
  const float* ls2      = (const float*)d_in[15];
  const int* bias_idxs  = (const int*)d_in[16];
  float* out = (float*)d_out;

  char* ws = (char*)d_ws;
  size_t off = 0;
  auto carve = [&](size_t bytes) {
    char* p = ws + off;
    off += (bytes + 255) & ~(size_t)255;
    return p;
  };
  __bf16* h      = (__bf16*)carve((size_t)Mrows * C * 2);   // LN1 out (reused for LN2)
  __bf16* WqkvT  = (__bf16*)carve((size_t)HQKV * C * 2);
  __bf16* WprojT = (__bf16*)carve((size_t)C * DH * 2);
  __bf16* W1T    = (__bf16*)carve((size_t)MH * C * 2);
  __bf16* W2T    = (__bf16*)carve((size_t)C * MH * 2);
  float* bias_full = (float*)carve((size_t)NH * NPAD * NPAD * 4);
  float* x1      = (float*)carve((size_t)Mrows * C * 4);    // residual after attn
  // aliased region: {q, k, vT, o} during attention; MLP hidden afterwards
  char* blk = ws + off;
  size_t qk_b = (size_t)Bn * NH * NPAD * KD * 2;   // 33.5 MB each
  size_t vt_b = (size_t)Bn * NH * DV * NPAD * 2;   // 134 MB
  __bf16* qB  = (__bf16*)blk;
  __bf16* kB  = (__bf16*)(blk + qk_b);
  __bf16* vT  = (__bf16*)(blk + 2 * qk_b);
  __bf16* oB  = (__bf16*)(blk + 2 * qk_b + vt_b);
  __bf16* hid = (__bf16*)blk;   // reuses q/k/vT(+o) after attention is consumed
  __bf16* h2  = h;

  // 1. weights -> bf16, K-major (tiny; stays hot in 192 MB L2)
  castT_kernel<<<(C * HQKV + 255) / 256, 256, 0, stream>>>(W_qkv, WqkvT, C, HQKV);
  castT_kernel<<<(DH * C + 255) / 256, 256, 0, stream>>>(W_proj, WprojT, DH, C);
  castT_kernel<<<(C * MH + 255) / 256, 256, 0, stream>>>(W1, W1T, C, MH);
  castT_kernel<<<(MH * C + 255) / 256, 256, 0, stream>>>(W2, W2T, MH, C);
  // 1b. expand + mask relative-position bias into fragment-order table
  bias_expand_kernel<<<(NH * NPAD * NPAD + 255) / 256, 256, 0, stream>>>(
      bias_tab, bias_idxs, bias_full);
  // 2. zero q/k/vT so token padding (49->64) contributes exact zeros
  int nz = (int)((2 * qk_b + vt_b) / 16);
  zero_kernel<<<(nz + 255) / 256, 256, 0, stream>>>((uint4*)blk, nz);
  // 3. LN1 -> bf16
  ln_cast_kernel<<<Mrows / 8, 256, 0, stream>>>(x, ln1_g, ln1_b, h);
  // 4. QKV GEMM (50176 x 512 x 1536), scatter epilogue
  wmma_gemm<C, EpiQKV><<<dim3(Mrows / 128, HQKV / 64), 256, 0, stream>>>(
      h, WqkvT, EpiQKV{b_qkv, qB, kB, vT});
  // 5. fused attention per (b,h)
  attn_kernel<<<Bn * NH, 128, 0, stream>>>(qB, kB, vT, bias_full, oB);
  // 6. proj GEMM + residual*ls1 -> x1 (f32)
  wmma_gemm<DH, EpiProj><<<dim3(Mrows / 128, C / 64), 256, 0, stream>>>(
      oB, WprojT, EpiProj{b_proj, ls1, x, x1});
  // 7. LN2 -> bf16
  ln_cast_kernel<<<Mrows / 8, 256, 0, stream>>>(x1, ln2_g, ln2_b, h2);
  // 8. MLP up-proj + exact GELU -> bf16 hidden
  wmma_gemm<C, EpiGelu><<<dim3(Mrows / 128, MH / 64), 256, 0, stream>>>(
      h2, W1T, EpiGelu{b1, hid});
  // 9. MLP down-proj + residual*ls2 -> out (f32)
  wmma_gemm<MH, EpiOut><<<dim3(Mrows / 128, C / 64), 256, 0, stream>>>(
      hid, W2T, EpiOut{b2, ls2, x1, out});
}
